// TemplateExtractor_65747359367635
// MI455X (gfx1250) — matrix operand order repaired and checked by reading, not compiled, not run on hardware
//
#include <hip/hip_runtime.h>
#include <hip/hip_bf16.h>

// ---------------------------------------------------------------------------
// TemplateExtractor on MI455X (gfx1250, wave32, WMMA)
//
// Roofline: ~5.1 GFLOP of 3x3-conv work vs ~10 MB of touched bytes -> matrix
// compute bound. Both convs run through V_WMMA_F32_16X16X32_F16 (f16 in, f32
// accumulate) as implicit GEMM: M = 49 output positions (4 wave tiles of 16),
// N = 256 output channels (16 tiles of 16), K = Cin*9.
// Input tiles are staged to LDS with gfx1250 async loads (ASYNCcnt path).
// ---------------------------------------------------------------------------

typedef _Float16 v8h  __attribute__((ext_vector_type(8)));
typedef _Float16 v16h __attribute__((ext_vector_type(16)));
typedef float    v8f  __attribute__((ext_vector_type(8)));

#define Bsz 32
#define Cin1 448
#define Hn 64
#define Wn 64
#define Tt 7
#define OC 256
#define PAD 9          // padded crop: 9x9 (7x7 interior + 1 halo for SAME conv)
#define NPOS 49        // 7*7 output positions
#define NK1 126        // 448*9/32 K-blocks for conv1
#define NK2 72         // 256*9/32 K-blocks for conv2

// ---- workspace layout (bytes) ----
#define OFF_CXY 0
#define OFF_XP  256
#define OFF_Y1  (OFF_XP + Bsz*81*Cin1*2)          // 2322688
#define OFF_W1  (OFF_Y1 + Bsz*81*OC*2)            // 3649792
#define OFF_W2  (OFF_W1 + OC*Cin1*9*2)            // 5714176

// ---------------------------------------------------------------------------
__global__ void zero_kernel(unsigned int* __restrict__ p, int n) {
  int i = blockIdx.x * blockDim.x + threadIdx.x;
  if (i < n) p[i] = 0u;
}

// ---------------------------------------------------------------------------
// Soft-argmax centroid: one block per batch. cy = sum(m*y)/(sum(m)+1e-8).
__global__ void centroid_kernel(const float* __restrict__ m, float* __restrict__ cxy) {
  __shared__ float rs[256], ry[256], rx[256];
  int b = blockIdx.x;
  const float* mb = m + (size_t)b * (Hn * Wn);
  float s = 0.f, sy = 0.f, sx = 0.f;
  for (int i = threadIdx.x; i < Hn * Wn; i += 256) {
    float v = mb[i];
    int y = i >> 6, x = i & 63;
    s += v; sy += v * (float)y; sx += v * (float)x;
  }
  int t = threadIdx.x;
  rs[t] = s; ry[t] = sy; rx[t] = sx;
  __syncthreads();
  for (int off = 128; off > 0; off >>= 1) {
    if (t < off) { rs[t] += rs[t+off]; ry[t] += ry[t+off]; rx[t] += rx[t+off]; }
    __syncthreads();
  }
  if (t == 0) {
    float inv = 1.f / (rs[0] + 1e-8f);
    cxy[b]       = ry[0] * inv;   // cy
    cxy[Bsz + b] = rx[0] * inv;   // cx
  }
}

// ---------------------------------------------------------------------------
// Repack OIHW fp32 weights into the exact B-fragment stream for
// v_wmma_f32_16x16x32_f16 (ISA 7.12.2): for K-block kblk, lane<16 holds
// K = kblk*32 + {0..7, 16..23} of column oc = oct*16+lane; lane>=16 holds {8..15, 24..31}.
// Each lane then loads 16 contiguous f16 (32B) per K-block.
__global__ void repack_kernel(const float* __restrict__ W, _Float16* __restrict__ Wp,
                              int Cin, int NK32) {
  int total = OC * Cin * 9;   // == 16 * NK32 * 32 * 16
  for (int idx = blockIdx.x * blockDim.x + threadIdx.x; idx < total;
       idx += gridDim.x * blockDim.x) {
    int e    = idx & 15;
    int lane = (idx >> 4) & 31;
    int kblk = (idx >> 9) % NK32;
    int oct  = (idx >> 9) / NK32;
    int oc   = oct * 16 + (lane & 15);
    int K    = kblk * 32 + ((lane < 16) ? 0 : 8) + (e < 8 ? e : e + 8);   // K = tap*Cin + ic
    int tap  = K / Cin;
    int ic   = K % Cin;
    int ky   = tap / 3, kx = tap % 3;
    Wp[idx] = (_Float16)W[(((size_t)oc * Cin + ic) * 3 + ky) * 3 + kx];
  }
}

// ---------------------------------------------------------------------------
// Bilinear 7x7 crop (align_corners pixel coords, zero padding) ->
// channel-last padded f16 tile Xp[b][(ty+1)*9+(tx+1)][c]. Border pre-zeroed.
__global__ void crop_kernel(const float* __restrict__ f, const float* __restrict__ cxy,
                            _Float16* __restrict__ Xp) {
  int b = blockIdx.y;
  int p = blockIdx.x;             // 0..48
  int ty = p / Tt, tx = p % Tt;
  float cy = cxy[b], cx = cxy[Bsz + b];
  float yy = cy - 3.0f + (float)ty;
  float xx = cx - 3.0f + (float)tx;
  float y0f = floorf(yy), x0f = floorf(xx);
  float wy1 = yy - y0f, wx1 = xx - x0f;
  float wy0 = 1.f - wy1, wx0 = 1.f - wx1;
  float y1f = y0f + 1.f, x1f = x0f + 1.f;
  bool vy0 = (y0f >= 0.f) && (y0f <= 63.f);
  bool vy1 = (y1f >= 0.f) && (y1f <= 63.f);
  bool vx0 = (x0f >= 0.f) && (x0f <= 63.f);
  bool vx1 = (x1f >= 0.f) && (x1f <= 63.f);
  int iy0 = (int)fminf(fmaxf(y0f, 0.f), 63.f);
  int iy1 = (int)fminf(fmaxf(y1f, 0.f), 63.f);
  int ix0 = (int)fminf(fmaxf(x0f, 0.f), 63.f);
  int ix1 = (int)fminf(fmaxf(x1f, 0.f), 63.f);
  float w00 = (vy0 && vx0) ? wy0 * wx0 : 0.f;
  float w01 = (vy0 && vx1) ? wy0 * wx1 : 0.f;
  float w10 = (vy1 && vx0) ? wy1 * wx0 : 0.f;
  float w11 = (vy1 && vx1) ? wy1 * wx1 : 0.f;
  int i00 = iy0 * Wn + ix0, i01 = iy0 * Wn + ix1;
  int i10 = iy1 * Wn + ix0, i11 = iy1 * Wn + ix1;
  const float* fb = f + (size_t)b * Cin1 * (Hn * Wn);
  _Float16* xrow = Xp + ((size_t)b * 81 + (ty + 1) * PAD + (tx + 1)) * Cin1;
  for (int c = threadIdx.x; c < Cin1; c += blockDim.x) {
    const float* fc = fb + (size_t)c * (Hn * Wn);
    float v = w00 * fc[i00] + w01 * fc[i01] + w10 * fc[i10] + w11 * fc[i11];
    xrow[c] = (_Float16)v;
  }
}

// ---------------------------------------------------------------------------
// Implicit-GEMM 3x3 conv over the 7x7 tile via v_wmma_f32_16x16x32_f16.
// Block = (one batch) x (one 16-channel output tile). 4 waves, each owning a
// 16-position M-tile. Input tile (padded 9x9 x CIN, f16, channel-last) is
// staged straight into LDS with gfx1250 async loads (cache -> LDS, ASYNCcnt).
// MODE 0: +bias, ReLU, f16 -> padded Y1p. MODE 1: +bias, f32 -> (B,256,7,7).
template <int CIN, int MODE>
__global__ void __launch_bounds__(128)
conv_wmma_kernel(const _Float16* __restrict__ Xin, const _Float16* __restrict__ Wp,
                 const float* __restrict__ bias,
                 _Float16* __restrict__ Yout, float* __restrict__ Fout) {
  constexpr int NK32 = CIN * 9 / 32;
  constexpr int NBLK = CIN / 32;
  extern __shared__ _Float16 lds[];     // only LDS object -> wave LDS offset 0
  int oct = blockIdx.x;                 // 0..15 output-channel tile
  int b   = blockIdx.y;                 // batch

  // Stage input tile to LDS via GLOBAL_LOAD_ASYNC_TO_LDS_B128 (ASYNCcnt path):
  // data flows cache->LDS without touching VGPRs.
  {
    const _Float16* src = Xin + (size_t)b * 81 * CIN;
    constexpr int n16 = (81 * CIN) >> 3;          // # of 16-byte chunks
    for (int i = threadIdx.x; i < n16; i += 128) {
      unsigned ldsOff = (unsigned)(i * 16);       // dynamic-LDS base offset 0
      const void* g = (const void*)(src + i * 8);
      asm volatile("global_load_async_to_lds_b128 %0, %1, off"
                   :: "v"(ldsOff), "v"(g)
                   : "memory");
    }
    asm volatile("s_wait_asynccnt 0" ::: "memory");  // my async ops done
  }
  __syncthreads();                                   // publish LDS block-wide

  int lane   = threadIdx.x & 31;
  int waveId = threadIdx.x >> 5;
  int m0     = waveId * 16;                 // M-tile base position
  int mrow   = lane & 15;
  int pos    = m0 + mrow;
  int posc   = pos > 48 ? 48 : pos;         // clamp padded rows (discarded on store)
  int oy     = posc / Tt, ox = posc % Tt;
  int kb     = (lane < 16) ? 0 : 8;         // A-fragment K sub-offset (ISA layout)

  v8f acc = {};
  int kblk = 0;
  for (int tap = 0; tap < 9; ++tap) {
    int ky = tap / 3, kx = tap % 3;
    const _Float16* xrow = &lds[((oy + ky) * PAD + (ox + kx)) * CIN];
#pragma unroll
    for (int c = 0; c < NBLK; ++c, ++kblk) {
      int ic0 = c << 5;
      // A fragment: lane<16 -> K {0..7, 16..23}; lane>=16 -> K {8..15, 24..31}
      v8h alo = *(const v8h*)(xrow + ic0 + kb);
      v8h ahi = *(const v8h*)(xrow + ic0 + kb + 16);
      v16h A = __builtin_shufflevector(alo, ahi,
                 0, 1, 2, 3, 4, 5, 6, 7, 8, 9, 10, 11, 12, 13, 14, 15);
      // B fragment: one contiguous 32B per lane from the repacked stream
      v16h Bf = *(const v16h*)(Wp + (((size_t)(oct * NK32 + kblk)) * 32 + lane) * 16);
      acc = __builtin_amdgcn_wmma_f32_16x16x32_f16(
              false, A, false, Bf, (short)0, acc, false, false);
    }
  }

  // D layout: element r, lane<16 -> (M=r, N=lane); lane>=16 -> (M=r+8, N=lane-16)
  int ocl  = oct * 16 + (lane & 15);
  float bv = bias[ocl];
  int madd = (lane < 16) ? 0 : 8;
#pragma unroll
  for (int r = 0; r < 8; ++r) {
    int p = m0 + r + madd;
    if (p < NPOS) {
      float v = acc[r] + bv;
      if (MODE == 0) {
        v = v > 0.f ? v : 0.f;
        int oyp = p / Tt, oxp = p % Tt;
        Yout[((size_t)b * 81 + (oyp + 1) * PAD + (oxp + 1)) * OC + ocl] = (_Float16)v;
      } else {
        Fout[((size_t)b * OC + ocl) * NPOS + p] = v;
      }
    }
  }
}

// ---------------------------------------------------------------------------
extern "C" void kernel_launch(void* const* d_in, const int* in_sizes, int n_in,
                              void* d_out, int out_size, void* d_ws, size_t ws_size,
                              hipStream_t stream) {
  const float* f_target = (const float*)d_in[0];
  const float* m_a      = (const float*)d_in[1];
  const float* W1       = (const float*)d_in[2];
  const float* b1       = (const float*)d_in[3];
  const float* W2       = (const float*)d_in[4];
  const float* b2       = (const float*)d_in[5];
  float* out = (float*)d_out;
  char* ws = (char*)d_ws;

  float*    cxy = (float*)(ws + OFF_CXY);
  _Float16* Xp  = (_Float16*)(ws + OFF_XP);
  _Float16* Y1  = (_Float16*)(ws + OFF_Y1);
  _Float16* Wp1 = (_Float16*)(ws + OFF_W1);
  _Float16* Wp2 = (_Float16*)(ws + OFF_W2);

  // zero Xp + Y1p (contiguous region) so padded borders are 0 every call
  int zeroDw = (OFF_W1 - OFF_XP) / 4;
  zero_kernel<<<(zeroDw + 255) / 256, 256, 0, stream>>>(
      (unsigned int*)(ws + OFF_XP), zeroDw);

  centroid_kernel<<<Bsz, 256, 0, stream>>>(m_a, cxy);

  repack_kernel<<<(OC * Cin1 * 9 + 255) / 256, 256, 0, stream>>>(W1, Wp1, Cin1, NK1);
  repack_kernel<<<(OC * OC   * 9 + 255) / 256, 256, 0, stream>>>(W2, Wp2, OC,   NK2);

  crop_kernel<<<dim3(NPOS, Bsz), 256, 0, stream>>>(f_target, cxy, Xp);

  conv_wmma_kernel<Cin1, 0><<<dim3(16, Bsz), 128, 81 * Cin1 * 2, stream>>>(
      Xp, Wp1, b1, Y1, nullptr);
  conv_wmma_kernel<OC, 1><<<dim3(16, Bsz), 128, 81 * OC * 2, stream>>>(
      Y1, Wp2, b2, nullptr, out);
}
